// SelfAttention_78520592105653
// MI455X (gfx1250) — compile-verified, matrix operand order checked by reading, not run on hardware
//
#include <hip/hip_runtime.h>
#include <hip/hip_bf16.h>

#define Bc 8
#define Tc 1024
#define Dc 1024
#define Hc 128
#define NSPLIT 4
#define VSPAN (Tc / NSPLIT)   // 256 keys per split

typedef __attribute__((ext_vector_type(16))) __bf16 v16bf;
typedef __attribute__((ext_vector_type(8)))  float  v8f;
typedef __attribute__((ext_vector_type(4)))  int    v4i;

union FragBF { v16bf v; unsigned u[8]; };

#if __has_builtin(__builtin_amdgcn_global_load_async_to_lds_b128) && \
    __has_builtin(__builtin_amdgcn_s_wait_asynccnt)
#define HAVE_ASYNC 1
#else
#define HAVE_ASYNC 0
#endif

#if __has_builtin(__builtin_amdgcn_cvt_pk_bf16_f32)
#define HAVE_CVTPK 1
#else
#define HAVE_CVTPK 0
#endif

// RNE single conversion (used on low-volume paths: q/k/v/W/P stores)
__device__ __forceinline__ unsigned short bf16_bits(float f) {
  unsigned u = __builtin_bit_cast(unsigned, f);
  return (unsigned short)((u + (0x7FFFu + ((u >> 16) & 1u))) >> 16);  // RNE
}
// RNE pair pack
__device__ __forceinline__ unsigned pk_bf16(float a, float b) {
#if HAVE_CVTPK
  typedef __attribute__((ext_vector_type(2))) __bf16 v2bf;
  v2bf r = __builtin_amdgcn_cvt_pk_bf16_f32(a, b);
  return __builtin_bit_cast(unsigned, r);
#else
  unsigned ua = __builtin_bit_cast(unsigned, a);
  unsigned ub = __builtin_bit_cast(unsigned, b);
  ua = (ua + (0x7FFFu + ((ua >> 16) & 1u))) >> 16;
  ub = (ub + (0x7FFFu + ((ub >> 16) & 1u)));
  return (ua & 0xFFFFu) | (ub & 0xFFFF0000u);
#endif
}
// Truncating pair pack: ONE v_perm_b32 — used on the hot streaming paths
__device__ __forceinline__ unsigned pk_bf16_tr(float a, float b) {
  unsigned ua = __builtin_bit_cast(unsigned, a);
  unsigned ub = __builtin_bit_cast(unsigned, b);
#if __has_builtin(__builtin_amdgcn_perm)
  return __builtin_amdgcn_perm(ub, ua, 0x07060302u);  // [b.hi16 | a.hi16]
#else
  return (ua >> 16) | (ub & 0xFFFF0000u);
#endif
}
__device__ __forceinline__ v8f wmma_bf16(const FragBF& a, const FragBF& b, v8f c) {
  return __builtin_amdgcn_wmma_f32_16x16x32_bf16(false, a.v, false, b.v,
                                                 (short)0, c, false, false);
}

// ---------------------------------------------------------------------------
// Kernel 1: fused QKV projection.  [B*T, D] @ [D, H] + bias -> bf16 q/k/v.
// grid = (64, 3): x = 128-row M tile, y selects {q,k,v}. 256 threads = 8 waves.
// x tile staged raw (f32) via async-to-LDS; bf16 conversion at fragment build.
// ---------------------------------------------------------------------------
__global__ __launch_bounds__(256) void qkv_proj_kernel(
    const float* __restrict__ x,
    const float* __restrict__ Wq, const float* __restrict__ bq,
    const float* __restrict__ Wk, const float* __restrict__ bk,
    const float* __restrict__ Wv, const float* __restrict__ bv,
    unsigned short* __restrict__ qbf, unsigned short* __restrict__ kbf,
    unsigned short* __restrict__ vbf)
{
  const int which = blockIdx.y;
  const float* W    = (which == 0) ? Wq : (which == 1) ? Wk : Wv;
  const float* bias = (which == 0) ? bq : (which == 1) ? bk : bv;
  unsigned short* out = (which == 0) ? qbf : (which == 1) ? kbf : vbf;

  __shared__ __align__(16) float          Axf[128][32];   // x tile, raw f32
  __shared__ __align__(16) unsigned short Wt[128][32];    // W tile transposed [n][k]

  const int tid  = threadIdx.x;
  const int lane = tid & 31;
  const int wv   = tid >> 5;        // wave 0..7 -> 16-row M sub-tile
  const int half = lane >> 4;
  const int l16  = lane & 15;
  const int m0   = blockIdx.x * 128;

  v8f acc[8] = {};                  // 8 N-tiles of 16 cols, f32 C

  for (int k0 = 0; k0 < Dc; k0 += 32) {
    { // stage x[m0..+128, k0..+32] f32 -> LDS (raw async copy)
      int r  = tid >> 1;
      int cb = (tid & 1) * 16;
      const float* gp = x + (size_t)(m0 + r) * Dc + k0 + cb;
      float* lp = &Axf[r][cb];
#if HAVE_ASYNC
      #pragma unroll
      for (int q = 0; q < 4; q++)
        __builtin_amdgcn_global_load_async_to_lds_b128(
            (__attribute__((address_space(1))) v4i*)(gp + 4*q),
            (__attribute__((address_space(3))) v4i*)(lp + 4*q), 0, 0);
#else
      #pragma unroll
      for (int q = 0; q < 4; q++)
        ((float4*)lp)[q] = ((const float4*)gp)[q];
#endif
    }
    { // load W[k0..+32, 0..128], convert bf16 (RNE), store transposed Wt[n][k]
      int kk = tid >> 3;
      int nb = (tid & 7) * 16;
      const float4* wp = (const float4*)(W + (size_t)(k0 + kk) * Hc + nb);
      #pragma unroll
      for (int q = 0; q < 4; q++) {
        float4 f = wp[q];
        int n = nb + 4*q;
        unsigned pxy = pk_bf16(f.x, f.y);
        unsigned pzw = pk_bf16(f.z, f.w);
        Wt[n+0][kk] = (unsigned short)pxy;
        Wt[n+1][kk] = (unsigned short)(pxy >> 16);
        Wt[n+2][kk] = (unsigned short)pzw;
        Wt[n+3][kk] = (unsigned short)(pzw >> 16);
      }
    }
#if HAVE_ASYNC
    __builtin_amdgcn_s_wait_asynccnt(0);
#endif
    __syncthreads();

    // A fragment: convert f32 pairs -> packed bf16 (reused by all 8 WMMAs)
    FragBF a;
    { int r = 16*wv + l16;                       // A row = lane%16
      const float2* ap = (const float2*)&Axf[r][0];
      #pragma unroll
      for (int j = 0; j < 8; j++) {
        float2 f = ap[j + half*8];
        a.u[j] = pk_bf16_tr(f.x, f.y);
      }
    }
    // preload all 8 B fragments, then issue WMMAs back-to-back
    FragBF bf[8];
    #pragma unroll
    for (int nt = 0; nt < 8; nt++) {
      const unsigned* wr = (const unsigned*)&Wt[nt*16 + l16][0];
      #pragma unroll
      for (int j = 0; j < 8; j++) bf[nt].u[j] = wr[j + half*8];
    }
    #pragma unroll
    for (int nt = 0; nt < 8; nt++)
      acc[nt] = wmma_bf16(a, bf[nt], acc[nt]);
    __syncthreads();
  }

  // bias + store bf16
  #pragma unroll
  for (int nt = 0; nt < 8; nt++) {
    int n = nt*16 + l16;
    float bs = bias[n];
    #pragma unroll
    for (int i = 0; i < 8; i++) {
      int m = m0 + 16*wv + i + 8*half;           // C row = i + 8*(lane>=16)
      out[(size_t)m * Hc + n] = bf16_bits(acc[nt][i] + bs);
    }
  }
}

// ---------------------------------------------------------------------------
// Kernel 2: fused flash attention partial.  grid = (T/16, NSPLIT).
// One block = one 16-row t-tile x ALL 8 batches (wave w <-> batch w), so
// rel_pos[t_tile, :, :] is streamed from HBM exactly once overall.
// ---------------------------------------------------------------------------
__global__ __launch_bounds__(256) void attn_partial_kernel(
    const unsigned short* __restrict__ qbf,
    const unsigned short* __restrict__ kbf,
    const unsigned short* __restrict__ vbf,
    const float* __restrict__ rel,
    float* __restrict__ pO, float* __restrict__ pM, float* __restrict__ pL)
{
  __shared__ __align__(16) float          rel_s[16][32][8];  // [t_local][v_local][batch]
  __shared__ __align__(16) unsigned short p_s[8][16][32];    // per-wave P tile (A layout)

  const int tid  = threadIdx.x;
  const int lane = tid & 31;
  const int wv   = tid >> 5;           // wave id == batch id
  const int half = lane >> 4;
  const int l16  = lane & 15;
  const int t0   = blockIdx.x * 16;
  const int vlo  = blockIdx.y * VSPAN;
  const int b    = wv;

  const unsigned* q32 = (const unsigned*)qbf;
  const unsigned* k32 = (const unsigned*)kbf;
  const float2*   rl2 = (const float2*)rel;

  v8f O[8] = {};                        // [h-tile][row i], rows t_local = i+8*half
  float m_i[8], l_i[8];
  #pragma unroll
  for (int i = 0; i < 8; i++) { m_i[i] = -1e30f; l_i[i] = 0.f; }

  const float scale = 11.313708498984761f;   // sqrt(H) per reference

  for (int v0 = vlo; v0 < vlo + VSPAN; v0 += 32) {
    // prefetch next rel_pos chunk for this wave's t rows
    if (v0 + 32 < vlo + VSPAN) {
      const char* pf = (const char*)(rel + ((size_t)(t0 + 2*wv) * Tc + (v0 + 32)) * Hc)
                       + (size_t)lane * 128;
      __builtin_prefetch(pf, 0, 0);
    }

    // ---- Phase A: rel scores (M = batch). wave w handles t_local {2w, 2w+1}.
    #pragma unroll
    for (int tt = 0; tt < 2; tt++) {
      int tl = 2*wv + tt;
      int t  = t0 + tl;
      v8f racc[2] = {};
      #pragma unroll
      for (int c0 = 0; c0 < Hc; c0 += 32) {
        FragBF a;
        int mb = l16;                               // A row = batch index
        if (mb < 8) {
          size_t base = (((size_t)mb * Tc + t) * Hc + c0 + half*16) >> 1;
          #pragma unroll
          for (int j = 0; j < 8; j++) a.u[j] = q32[base + j];
        } else {
          #pragma unroll
          for (int j = 0; j < 8; j++) a.u[j] = 0u;
        }
        #pragma unroll
        for (int nt = 0; nt < 2; nt++) {
          FragBF bb;
          int vv = v0 + nt*16 + l16;
          size_t rb = (((size_t)t * Tc + vv) * Hc + c0 + half*16) >> 1;  // float2 idx
          #pragma unroll
          for (int j = 0; j < 8; j++) {
            float2 f = rl2[rb + j];
            bb.u[j] = pk_bf16_tr(f.x, f.y);        // 1x v_perm_b32
          }
          racc[nt] = wmma_bf16(a, bb, racc[nt]);
        }
      }
      if (lane < 16) {                 // valid batches live in VGPRs 0..7, lanes 0..15
        #pragma unroll
        for (int nt = 0; nt < 2; nt++)
          #pragma unroll
          for (int i = 0; i < 8; i++)
            rel_s[tl][nt*16 + lane][i] = racc[nt][i];
      }
    }
    __syncthreads();

    // ---- Phase B: scores = scale * q k^T + rel (wave w == batch w)
    v8f S[2] = {};
    #pragma unroll
    for (int c0 = 0; c0 < Hc; c0 += 32) {
      FragBF a;
      { size_t base = (((size_t)b * Tc + (t0 + l16)) * Hc + c0 + half*16) >> 1;
        #pragma unroll
        for (int j = 0; j < 8; j++) a.u[j] = q32[base + j];
      }
      FragBF bk0, bk1;
      { size_t base = (((size_t)b * Tc + (v0 + l16)) * Hc + c0 + half*16) >> 1;
        #pragma unroll
        for (int j = 0; j < 8; j++) bk0.u[j] = k32[base + j];
      }
      { size_t base = (((size_t)b * Tc + (v0 + 16 + l16)) * Hc + c0 + half*16) >> 1;
        #pragma unroll
        for (int j = 0; j < 8; j++) bk1.u[j] = k32[base + j];
      }
      S[0] = wmma_bf16(a, bk0, S[0]);
      S[1] = wmma_bf16(a, bk1, S[1]);
    }
    #pragma unroll
    for (int nt = 0; nt < 2; nt++)
      #pragma unroll
      for (int i = 0; i < 8; i++)
        S[nt][i] = S[nt][i] * scale + rel_s[i + 8*half][nt*16 + l16][b];

    // ---- Phase C: online softmax (rows spread over 16-lane halves)
    float mc[8];
    #pragma unroll
    for (int i = 0; i < 8; i++) mc[i] = fmaxf(S[0][i], S[1][i]);
    #pragma unroll
    for (int d = 1; d < 16; d <<= 1)
      #pragma unroll
      for (int i = 0; i < 8; i++) mc[i] = fmaxf(mc[i], __shfl_xor(mc[i], d, 32));

    float alpha[8];
    #pragma unroll
    for (int i = 0; i < 8; i++) {
      float mn = fmaxf(m_i[i], mc[i]);
      alpha[i] = __expf(m_i[i] - mn);
      m_i[i] = mn;
    }
    float P0[8], P1[8], rs[8];
    #pragma unroll
    for (int i = 0; i < 8; i++) {
      P0[i] = __expf(S[0][i] - m_i[i]);
      P1[i] = __expf(S[1][i] - m_i[i]);
      rs[i] = P0[i] + P1[i];
    }
    #pragma unroll
    for (int d = 1; d < 16; d <<= 1)
      #pragma unroll
      for (int i = 0; i < 8; i++) rs[i] += __shfl_xor(rs[i], d, 32);
    #pragma unroll
    for (int i = 0; i < 8; i++) l_i[i] = l_i[i] * alpha[i] + rs[i];
    #pragma unroll
    for (int nt = 0; nt < 8; nt++)
      #pragma unroll
      for (int i = 0; i < 8; i++)
        O[nt][i] *= alpha[i];

    // ---- Phase D: C-layout P -> A-layout via per-wave LDS tile
    #pragma unroll
    for (int i = 0; i < 8; i++) {
      p_s[wv][i + 8*half][ 0 + l16] = bf16_bits(P0[i]);
      p_s[wv][i + 8*half][16 + l16] = bf16_bits(P1[i]);
    }
    __syncthreads();

    // ---- Phase E: O += P @ V   (preload all 8 B fragments, then 8 WMMAs)
    FragBF pa;
    { const unsigned* pr = (const unsigned*)&p_s[wv][l16][0];
      #pragma unroll
      for (int j = 0; j < 8; j++) pa.u[j] = pr[j + half*8];
    }
    const unsigned short* vb = vbf + (size_t)b * Tc * Hc;
    FragBF bb[8];
    #pragma unroll
    for (int nt = 0; nt < 8; nt++) {
      int h = nt*16 + l16;
      #pragma unroll
      for (int j = 0; j < 8; j++) {
        int kk = 2*j + half*16;
        unsigned lo = vb[(size_t)(v0 + kk    ) * Hc + h];
        unsigned hi = vb[(size_t)(v0 + kk + 1) * Hc + h];
        bb[nt].u[j] = lo | (hi << 16);
      }
    }
    #pragma unroll
    for (int nt = 0; nt < 8; nt++)
      O[nt] = wmma_bf16(pa, bb[nt], O[nt]);
  }

  // ---- write partial O / m / l for this split
  size_t ob = ((size_t)blockIdx.y * Bc + b) * Tc * Hc;
  #pragma unroll
  for (int nt = 0; nt < 8; nt++) {
    int h = nt*16 + l16;
    #pragma unroll
    for (int i = 0; i < 8; i++) {
      int t = t0 + i + 8*half;
      pO[ob + (size_t)t * Hc + h] = O[nt][i];
    }
  }
  if (l16 == 0) {
    size_t mb = ((size_t)blockIdx.y * Bc + b) * Tc;
    #pragma unroll
    for (int i = 0; i < 8; i++) {
      int t = t0 + i + 8*half;
      pM[mb + t] = m_i[i];
      pL[mb + t] = l_i[i];
    }
  }
}

// ---------------------------------------------------------------------------
// Kernel 3: combine NSPLIT partials -> final output (f32 [B,T,H])
// ---------------------------------------------------------------------------
__global__ __launch_bounds__(256) void attn_combine_kernel(
    const float* __restrict__ pO, const float* __restrict__ pM,
    const float* __restrict__ pL, float* __restrict__ out)
{
  size_t idx = (size_t)blockIdx.x * 256 + threadIdx.x;   // over B*T*H
  size_t bt = idx / Hc;
  float m = -1e30f;
  #pragma unroll
  for (int s = 0; s < NSPLIT; s++)
    m = fmaxf(m, pM[(size_t)s * Bc * Tc + bt]);
  float lsum = 0.f, osum = 0.f;
  #pragma unroll
  for (int s = 0; s < NSPLIT; s++) {
    float w = __expf(pM[(size_t)s * Bc * Tc + bt] - m);
    lsum += w * pL[(size_t)s * Bc * Tc + bt];
    osum += w * pO[(size_t)s * Bc * Tc * Hc + idx];
  }
  out[idx] = osum / lsum;
}

// ---------------------------------------------------------------------------
extern "C" void kernel_launch(void* const* d_in, const int* in_sizes, int n_in,
                              void* d_out, int out_size, void* d_ws, size_t ws_size,
                              hipStream_t stream) {
  const float* x   = (const float*)d_in[0];
  const float* Wq  = (const float*)d_in[1];
  const float* bq  = (const float*)d_in[2];
  const float* Wk  = (const float*)d_in[3];
  const float* bk  = (const float*)d_in[4];
  const float* Wv  = (const float*)d_in[5];
  const float* bv  = (const float*)d_in[6];
  const float* rel = (const float*)d_in[7];
  float* out = (float*)d_out;

  char* ws = (char*)d_ws;
  const size_t nBTH = (size_t)Bc * Tc * Hc;
  unsigned short* qbf = (unsigned short*)ws;                 // 2 MB
  unsigned short* kbf = (unsigned short*)(ws + nBTH * 2);    // 2 MB
  unsigned short* vbf = (unsigned short*)(ws + nBTH * 4);    // 2 MB
  float* pO = (float*)(ws + nBTH * 6);                       // 16 MB
  float* pM = (float*)(ws + nBTH * 6 + (size_t)NSPLIT * nBTH * 4);
  float* pL = pM + (size_t)NSPLIT * Bc * Tc;

  qkv_proj_kernel<<<dim3((Bc * Tc) / 128, 3), 256, 0, stream>>>(
      x, Wq, bq, Wk, bk, Wv, bv, qbf, kbf, vbf);
  attn_partial_kernel<<<dim3(Tc / 16, NSPLIT), 256, 0, stream>>>(
      qbf, kbf, vbf, rel, pO, pM, pL);
  attn_combine_kernel<<<(unsigned)(nBTH / 256), 256, 0, stream>>>(pO, pM, pL, out);
}
